// WindowSelfAttention_21449066676781
// MI455X (gfx1250) — matrix operand order repaired and checked by reading, not compiled, not run on hardware
//
#include <hip/hip_runtime.h>
#include <hip/hip_bf16.h>

typedef _Float16 v16h __attribute__((ext_vector_type(16)));
typedef _Float16 v8h  __attribute__((ext_vector_type(8)));
typedef float    v8f  __attribute__((ext_vector_type(8)));

#define WIN_N  49
#define HEADS  6
#define HDIM   32
#define CDIM   192
#define NBATCH 4096
#define NROWS  (NBATCH * WIN_N)   // 200704

// ---------------------------------------------------------------------------
// WMMA + fragment helpers (CDNA5 v_wmma_f32_16x16x32_f16, wave32)
// ---------------------------------------------------------------------------
__device__ __forceinline__ v8f wmma16(v16h a, v16h b, v8f c) {
  return __builtin_amdgcn_wmma_f32_16x16x32_f16(false, a, false, b, (short)0, c,
                                                false, false);
}

// A fragment (16x32 f16), tile row-major [row][k], leading dim ld (halfs).
__device__ __forceinline__ v16h frag_a(const _Float16* p, int ld, int row0, int koff) {
  const int lane = threadIdx.x & 31;
  const int hf   = (lane >> 4) & 1;
  const int l16  = lane & 15;
  const _Float16* rp = p + (row0 + l16) * ld + koff + hf * 8;
  v8h lo = *(const v8h*)(rp);
  v8h hi = *(const v8h*)(rp + 16);
  return __builtin_shufflevector(lo, hi, 0, 1, 2, 3, 4, 5, 6, 7,
                                 8, 9, 10, 11, 12, 13, 14, 15);
}

// B fragment (32x16 f16), tile stored TRANSPOSED row-major [n][k].
__device__ __forceinline__ v16h frag_b(const _Float16* p, int ld, int col0, int koff) {
  const int lane = threadIdx.x & 31;
  const int hf   = (lane >> 4) & 1;
  const int l16  = lane & 15;
  const _Float16* rp = p + (col0 + l16) * ld + koff + hf * 8;
  v8h lo = *(const v8h*)(rp);
  v8h hi = *(const v8h*)(rp + 16);
  return __builtin_shufflevector(lo, hi, 0, 1, 2, 3, 4, 5, 6, 7,
                                 8, 9, 10, 11, 12, 13, 14, 15);
}

// ---------------------------------------------------------------------------
// CDNA5 async global->LDS copy, 16 bytes per active lane. lds_off is a byte
// offset into this workgroup's LDS allocation (dynamic-shared base obtained
// from __builtin_amdgcn_groupstaticsize()). Tracked by ASYNCcnt.
// ---------------------------------------------------------------------------
__device__ __forceinline__ void async_b128(unsigned lds_off, const void* gsrc) {
  asm volatile("global_load_async_to_lds_b128 %0, %1, off"
               :: "v"(lds_off), "v"(gsrc)
               : "memory");
}
__device__ __forceinline__ void wait_async0() {
  asm volatile("s_wait_asynccnt 0x0" ::: "memory");
}

// ---------------------------------------------------------------------------
// Kernel 1: QKV GEMM  [200704 x 192] x [192 x 576] + bias -> f16 q/k/v
// 256 threads (8 waves), 128x64 tile, double-buffered LDS, reg prefetch.
// q pre-scaled by D^-0.5. Sources are f32 so the convert goes through VGPRs.
// ---------------------------------------------------------------------------
__global__ __launch_bounds__(256) void qkv_gemm(
    const float* __restrict__ x, const float* __restrict__ w,
    const float* __restrict__ bias,
    _Float16* __restrict__ qb, _Float16* __restrict__ kb, _Float16* __restrict__ vb) {
  __shared__ __align__(16) _Float16 As[2][128][40];
  __shared__ __align__(16) _Float16 Bs[2][64][40];
  const int tid = threadIdx.x;
  const int bm = blockIdx.x, bn = blockIdx.y;
  const int wid = tid >> 5, wr = wid >> 1, wc = wid & 1;
  v8f acc[2][2] = {};

  float4 ar[4];
  float  br[8];

  auto load_regs = [&](int k0) {
#pragma unroll
    for (int it = 0; it < 4; ++it) {
      int i = it * 256 + tid;               // 0..1023
      int rA = i >> 3, c4 = i & 7;
      ar[it] = *(const float4*)(x + (size_t)(bm * 128 + rA) * CDIM + k0 + c4 * 4);
    }
#pragma unroll
    for (int it = 0; it < 8; ++it) {
      int i = it * 256 + tid;               // 0..2047
      int kk = i >> 6, n = i & 63;
      br[it] = w[(size_t)(k0 + kk) * (3 * CDIM) + bn * 64 + n];
    }
  };
  auto store_lds = [&](int buf) {
#pragma unroll
    for (int it = 0; it < 4; ++it) {
      int i = it * 256 + tid;
      int rA = i >> 3, c4 = i & 7;
      As[buf][rA][c4 * 4 + 0] = (_Float16)ar[it].x;
      As[buf][rA][c4 * 4 + 1] = (_Float16)ar[it].y;
      As[buf][rA][c4 * 4 + 2] = (_Float16)ar[it].z;
      As[buf][rA][c4 * 4 + 3] = (_Float16)ar[it].w;
    }
#pragma unroll
    for (int it = 0; it < 8; ++it) {
      int i = it * 256 + tid;
      int kk = i >> 6, n = i & 63;
      Bs[buf][n][kk] = (_Float16)br[it];
    }
  };

  load_regs(0);
  store_lds(0);
  __syncthreads();

#pragma unroll
  for (int kt = 0; kt < 6; ++kt) {
    if (kt < 5) {
      load_regs(32 * (kt + 1));
      if (kt < 4)   // warm L2/WGP$ two steps ahead (global_prefetch_b8)
        __builtin_prefetch(x + (size_t)(bm * 128 + (tid >> 1)) * CDIM + 32 * (kt + 2), 0, 0);
    }
    const _Float16* Ab = &As[kt & 1][0][0];
    const _Float16* Bb = &Bs[kt & 1][0][0];
    v16h a0 = frag_a(Ab, 40, wr * 32, 0);
    v16h a1 = frag_a(Ab, 40, wr * 32 + 16, 0);
    v16h b0 = frag_b(Bb, 40, wc * 32, 0);
    v16h b1 = frag_b(Bb, 40, wc * 32 + 16, 0);
    acc[0][0] = wmma16(a0, b0, acc[0][0]);
    acc[0][1] = wmma16(a0, b1, acc[0][1]);
    acc[1][0] = wmma16(a1, b0, acc[1][0]);
    acc[1][1] = wmma16(a1, b1, acc[1][1]);
    if (kt < 5) {
      store_lds((kt + 1) & 1);
      __syncthreads();
    }
  }

  const int lane = tid & 31, hf = (lane >> 4) & 1, l16 = lane & 15;
#pragma unroll
  for (int i = 0; i < 2; ++i)
#pragma unroll
    for (int j = 0; j < 2; ++j)
#pragma unroll
      for (int r = 0; r < 8; ++r) {
        int m = wr * 32 + i * 16 + hf * 8 + r;
        int gr = bm * 128 + m;
        int gc = bn * 64 + wc * 32 + j * 16 + l16;
        float val = acc[i][j][r] + bias[gc];
        int which = gc / CDIM;
        int c2 = gc - which * CDIM;
        int h = c2 >> 5, d = c2 & 31;
        if (which == 0) val *= 0.17677669529663687f;   // D^-0.5
        int b = gr / WIN_N;
        int tok = gr - b * WIN_N;
        _Float16* dst = (which == 0) ? qb : (which == 1) ? kb : vb;
        dst[((size_t)(b * HEADS + h) * WIN_N + tok) * HDIM + d] = (_Float16)val;
      }
}

// ---------------------------------------------------------------------------
// Kernel 2: fused attention per (b,h).  Dynamic LDS layout (halfs):
//   Qs  @    0  [64][40]      Ks  @ 2560  [64][40]
//   VT  @ 5120  [32][72]      Ps  @ 7424  [64][72]     total 12032 h = 24064 B
// Q/K tiles come in via async global->LDS B128; V is transposed manually.
// ---------------------------------------------------------------------------
#define K2_KS 2560
#define K2_VT 5120
#define K2_PS 7424

__global__ __launch_bounds__(128) void attn_fused(
    const _Float16* __restrict__ qb, const _Float16* __restrict__ kb,
    const _Float16* __restrict__ vb, const float* __restrict__ btab,
    _Float16* __restrict__ ot) {
  extern __shared__ __align__(16) _Float16 smem[];
  const unsigned lds_base = __builtin_amdgcn_groupstaticsize();
  const int bh = blockIdx.x;
  const int b = bh / HEADS, h = bh - b * HEADS;
  const _Float16* qg = qb + (size_t)bh * WIN_N * HDIM;
  const _Float16* kg = kb + (size_t)bh * WIN_N * HDIM;
  const _Float16* vg = vb + (size_t)bh * WIN_N * HDIM;
  const int tid = threadIdx.x;

  // Q/K: rows < 49 async-copied (4 x 16B per row), pad rows zero-filled.
  for (int idx = tid; idx < 256; idx += 128) {
    int r = idx >> 2, c = idx & 3;
    if (r < WIN_N) {
      async_b128(lds_base + (unsigned)(r * 80 + c * 16),          qg + r * 32 + c * 8);
      async_b128(lds_base + (unsigned)(2 * K2_KS + r * 80 + c * 16), kg + r * 32 + c * 8);
    } else {
      v8h z = {};
      *(v8h*)(smem + r * 40 + c * 8) = z;
      *(v8h*)(smem + K2_KS + r * 40 + c * 8) = z;
    }
  }
  // V transposed into VT[d][tok] with zero padding.
  for (int idx = tid; idx < 64 * 32; idx += 128) {
    int r = idx >> 5, c = idx & 31;
    smem[K2_VT + c * 72 + r] = (r < WIN_N) ? vg[idx] : (_Float16)0.0f;
  }
  wait_async0();
  __syncthreads();

  const int wv = tid >> 5;
  const int m0 = wv * 16;
  const int lane = tid & 31, hf = (lane >> 4) & 1, l16 = lane & 15;

  // S = q k^T  (K tile [tok][d] is already transposed-B layout)
  v16h aq = frag_a(smem, 40, m0, 0);
  v8f s[4];
#pragma unroll
  for (int j = 0; j < 4; ++j) {
    v16h bk = frag_b(smem + K2_KS, 40, j * 16, 0);
    v8f z = {};
    s[j] = wmma16(aq, bk, z);
  }

  // bias + mask + softmax
  float vals[4][8], rmax[8], rsum[8];
#pragma unroll
  for (int r = 0; r < 8; ++r) rmax[r] = -1e30f;
#pragma unroll
  for (int j = 0; j < 4; ++j)
#pragma unroll
    for (int r = 0; r < 8; ++r) {
      int m = m0 + hf * 8 + r;
      int n = j * 16 + l16;
      float val = -1e30f;
      if (n < WIN_N) {
        int mm = (m < WIN_N) ? m : (WIN_N - 1);
        int mh = mm / 7, mw = mm - mh * 7;
        int nh = n / 7,  nw = n - nh * 7;
        int rel = (mh - nh + 6) * 13 + (mw - nw + 6);
        val = s[j][r] + btab[rel * HEADS + h];
      }
      vals[j][r] = val;
      rmax[r] = fmaxf(rmax[r], val);
    }
#pragma unroll
  for (int r = 0; r < 8; ++r) {
#pragma unroll
    for (int off = 1; off < 16; off <<= 1)
      rmax[r] = fmaxf(rmax[r], __shfl_xor(rmax[r], off, 32));
    rsum[r] = 0.0f;
  }
#pragma unroll
  for (int j = 0; j < 4; ++j)
#pragma unroll
    for (int r = 0; r < 8; ++r) {
      float e = __expf(vals[j][r] - rmax[r]);
      rsum[r] += e;
      int m = m0 + hf * 8 + r;
      int n = j * 16 + l16;
      smem[K2_PS + m * 72 + n] = (_Float16)e;
    }
#pragma unroll
  for (int r = 0; r < 8; ++r)
#pragma unroll
    for (int off = 1; off < 16; off <<= 1)
      rsum[r] += __shfl_xor(rsum[r], off, 32);
  __syncthreads();

  // O = P V  (K = 64 padded -> 2 k-steps; 2 column tiles of d)
#pragma unroll
  for (int ct = 0; ct < 2; ++ct) {
    v8f o = {};
#pragma unroll
    for (int kk = 0; kk < 2; ++kk) {
      v16h ap = frag_a(smem + K2_PS, 72, m0, kk * 32);
      v16h bv = frag_b(smem + K2_VT, 72, ct * 16, kk * 32);
      o = wmma16(ap, bv, o);
    }
#pragma unroll
    for (int r = 0; r < 8; ++r) {
      int m = m0 + hf * 8 + r;
      if (m < WIN_N) {
        int d = ct * 16 + l16;
        float val = o[r] / rsum[r];
        ot[((size_t)(b * WIN_N + m)) * CDIM + h * HDIM + d] = (_Float16)val;
      }
    }
  }
}

// ---------------------------------------------------------------------------
// Kernel 3: output projection [200704 x 192] x [192 x 192] -> f32 out.
// Dynamic LDS (bytes): As buf0 @ 0, buf1 @ 10240; Bs buf0 @ 20480, buf1 @ 25600.
// A tile (already f16) streamed via async global->LDS; B converted via regs.
// Double-buffered, one barrier per K step.
// ---------------------------------------------------------------------------
#define K3_AS_H 5120     // halfs per A buffer (128*40)
#define K3_BS_H 2560     // halfs per B buffer (64*40)
#define K3_BS0_H 10240   // half offset of Bs region

__global__ __launch_bounds__(256) void proj_gemm(
    const _Float16* __restrict__ Ag, const float* __restrict__ W,
    float* __restrict__ out) {
  extern __shared__ __align__(16) _Float16 smem[];
  const unsigned lds_base = __builtin_amdgcn_groupstaticsize();
  const int tid = threadIdx.x;
  const int bm = blockIdx.x, bn = blockIdx.y;
  const int wid = tid >> 5, wr = wid >> 1, wc = wid & 1;
  v8f acc[2][2] = {};
  float br[8];

  auto issue_A = [&](int k0, int buf) {
#pragma unroll
    for (int it = 0; it < 2; ++it) {
      int i = it * 256 + tid;               // 0..511 (16B chunks)
      int rA = i >> 2, c = i & 3;
      unsigned loff = lds_base + (unsigned)(buf * (K3_AS_H * 2) + rA * 80 + c * 16);
      async_b128(loff, Ag + (size_t)(bm * 128 + rA) * CDIM + k0 + c * 8);
    }
  };
  auto load_B = [&](int k0) {
#pragma unroll
    for (int it = 0; it < 8; ++it) {
      int i = it * 256 + tid;               // 0..2047
      int kk = i >> 6, n = i & 63;
      br[it] = W[(size_t)(k0 + kk) * CDIM + bn * 64 + n];
    }
  };
  auto store_B = [&](int buf) {
#pragma unroll
    for (int it = 0; it < 8; ++it) {
      int i = it * 256 + tid;
      int kk = i >> 6, n = i & 63;
      smem[K3_BS0_H + buf * K3_BS_H + n * 40 + kk] = (_Float16)br[it];
    }
  };

  issue_A(0, 0);
  load_B(0);
  store_B(0);
  wait_async0();
  __syncthreads();

#pragma unroll
  for (int kt = 0; kt < 6; ++kt) {
    if (kt < 5) {
      issue_A(32 * (kt + 1), (kt + 1) & 1);
      load_B(32 * (kt + 1));
    }
    const _Float16* Ab = smem + (kt & 1) * K3_AS_H;
    const _Float16* Bb = smem + K3_BS0_H + (kt & 1) * K3_BS_H;
    v16h a0 = frag_a(Ab, 40, wr * 32, 0);
    v16h a1 = frag_a(Ab, 40, wr * 32 + 16, 0);
    v16h b0 = frag_b(Bb, 40, wc * 32, 0);
    v16h b1 = frag_b(Bb, 40, wc * 32 + 16, 0);
    acc[0][0] = wmma16(a0, b0, acc[0][0]);
    acc[0][1] = wmma16(a0, b1, acc[0][1]);
    acc[1][0] = wmma16(a1, b0, acc[1][0]);
    acc[1][1] = wmma16(a1, b1, acc[1][1]);
    if (kt < 5) {
      store_B((kt + 1) & 1);
      wait_async0();
      __syncthreads();
    }
  }

  const int lane = tid & 31, hf = (lane >> 4) & 1, l16 = lane & 15;
#pragma unroll
  for (int i = 0; i < 2; ++i)
#pragma unroll
    for (int j = 0; j < 2; ++j)
#pragma unroll
      for (int r = 0; r < 8; ++r) {
        int m = wr * 32 + i * 16 + hf * 8 + r;
        int gr = bm * 128 + m;
        int gc = bn * 64 + wc * 32 + j * 16 + l16;
        out[(size_t)gr * CDIM + gc] = acc[i][j][r];
      }
}

// ---------------------------------------------------------------------------
extern "C" void kernel_launch(void* const* d_in, const int* in_sizes, int n_in,
                              void* d_out, int out_size, void* d_ws, size_t ws_size,
                              hipStream_t stream) {
  const float* x      = (const float*)d_in[0];
  const float* w_qkv  = (const float*)d_in[1];
  const float* b_qkv  = (const float*)d_in[2];
  const float* btab   = (const float*)d_in[3];
  const float* w_proj = (const float*)d_in[4];
  float* out = (float*)d_out;

  const size_t qkv_elems = (size_t)NBATCH * HEADS * WIN_N * HDIM;  // 38,535,168
  _Float16* qb = (_Float16*)d_ws;
  _Float16* kb = qb + qkv_elems;
  _Float16* vb = kb + qkv_elems;
  _Float16* ot = vb + qkv_elems;   // [NROWS][CDIM] f16

  dim3 g1(NROWS / 128, (3 * CDIM) / 64);          // 1568 x 9
  qkv_gemm<<<g1, 256, 0, stream>>>(x, w_qkv, b_qkv, qb, kb, vb);

  dim3 g2(NBATCH * HEADS);                         // 24576
  attn_fused<<<g2, 128, 24064, stream>>>(qb, kb, vb, btab, ot);

  dim3 g3(NROWS / 128, CDIM / 64);                 // 1568 x 3
  proj_gemm<<<g3, 256, 30720, stream>>>(ot, w_proj, out);
}
